// Attention_83648783057736
// MI455X (gfx1250) — matrix operand order repaired and checked by reading, not compiled, not run on hardware
//
#include <hip/hip_runtime.h>
#include <hip/hip_bf16.h>

// ---------------------------------------------------------------------------
// CDNA5 (gfx1250, wave32) attention via V_WMMA_F32_16X16X32_BF16.
// All hot-loop operand fragments are contiguous 16B loads (bf16, pre-laid-out).
// ---------------------------------------------------------------------------

typedef __attribute__((ext_vector_type(16))) __bf16 v16bf;
typedef __attribute__((ext_vector_type(8)))  __bf16 v8bf;
typedef __attribute__((ext_vector_type(4)))  __bf16 v4bf;
typedef __attribute__((ext_vector_type(8)))  float  v8f;

#define DIM    512
#define HEADS  8
#define DHEAD  64
#define SEQ    2048
#define BATCH  4
#define INNER  (HEADS * DHEAD)      // 512
#define MTOT   (BATCH * SEQ)        // 8192
#define ATTN_SCALE 0.125f           // 64^-0.5

__device__ __forceinline__ v8f wmma_bf16(v16bf a, v16bf b, v8f c) {
  // (neg_a, A, neg_b, B, c_mod, C, reuse_a, reuse_b)
  return __builtin_amdgcn_wmma_f32_16x16x32_bf16(false, a, false, b,
                                                 (short)0, c, false, false);
}

// ---- Fragment loaders (wave32, contiguous k order shared by A and B) -------------------------
// A (16x32 bf16, row-major src): lane L -> row M=(L&15),
//   elems e = K(half*16+e).
__device__ __forceinline__ v16bf load_a_bf16(const __bf16* __restrict__ A,
                                             int ld, int m0, int k0, int lane) {
  int half = lane >> 4, r = lane & 15;
  const __bf16* row = A + (size_t)(m0 + r) * ld + k0 + half * 16;
  v8bf lo = *(const v8bf*)(row);
  v8bf hi = *(const v8bf*)(row + 8);
  v16bf a;
#pragma unroll
  for (int i = 0; i < 8; ++i) { a[i] = lo[i]; a[8 + i] = hi[i]; }
  return a;
}

// B (32x16 bf16) = transpose of row-major src[n][k] (e.g. W^T stored [N][K],
// or Q^T from Q[n][d]): per-lane 16 K-elements are CONTIGUOUS (2x16B loads).
// Layout: lane c -> col N=(c&15), elems e = K(half*16+e).
__device__ __forceinline__ v16bf load_bt_bf16(const __bf16* __restrict__ S,
                                              int ld, int k0, int n0, int lane) {
  int half = lane >> 4, c = lane & 15;
  const __bf16* p = S + (size_t)(n0 + c) * ld + k0 + half * 16;
  v8bf lo = *(const v8bf*)(p);
  v8bf hi = *(const v8bf*)(p + 8);
  v16bf b;
#pragma unroll
  for (int i = 0; i < 8; ++i) { b[i] = lo[i]; b[8 + i] = hi[i]; }
  return b;
}

// ---------------------------------------------------------------------------
// Prep kernels (run once, tiny cost): f32 -> bf16 convert, and weight
// transpose+convert W[K][N] f32 -> WT[N][K] bf16.
// ---------------------------------------------------------------------------
__global__ void cvt_bf16_kernel(const float* __restrict__ src,
                                __bf16* __restrict__ dst, int n4) {
  int t = blockIdx.x * blockDim.x + threadIdx.x;
  if (t >= n4) return;
  float4 v = *(const float4*)(src + (size_t)t * 4);
  v4bf o;
  o[0] = (__bf16)v.x; o[1] = (__bf16)v.y; o[2] = (__bf16)v.z; o[3] = (__bf16)v.w;
  *(v4bf*)(dst + (size_t)t * 4) = o;
}

__global__ void transpose_w_kernel(const float* __restrict__ W,  // [512][N]
                                   __bf16* __restrict__ WT,      // [N][512]
                                   int N) {
  int t = blockIdx.x * blockDim.x + threadIdx.x;
  if (t >= N * DIM) return;
  int k = t & (DIM - 1);
  int n = t >> 9;
  WT[(size_t)n * DIM + k] = (__bf16)W[(size_t)k * N + n];
}

// ---------------------------------------------------------------------------
// 32x64 register-blocked GEMM core: C[2][4] (16x16 tiles) over K=512.
// A bf16 row-major [M][512]; BT bf16 row-major [N][512] (transposed weights).
// ---------------------------------------------------------------------------
__device__ __forceinline__ void gemm_32x64(const __bf16* __restrict__ A,
                                           const __bf16* __restrict__ BT,
                                           int m0, int n0, int lane,
                                           v8f acc[2][4]) {
#pragma unroll 2
  for (int k0 = 0; k0 < DIM; k0 += 32) {
    v16bf a0 = load_a_bf16(A, DIM, m0, k0, lane);
    v16bf a1 = load_a_bf16(A, DIM, m0 + 16, k0, lane);
#pragma unroll
    for (int ni = 0; ni < 4; ++ni) {
      v16bf b = load_bt_bf16(BT, DIM, k0, n0 + 16 * ni, lane);
      acc[0][ni] = wmma_bf16(a0, b, acc[0][ni]);
      acc[1][ni] = wmma_bf16(a1, b, acc[1][ni]);
    }
  }
}

// ---------------------------------------------------------------------------
// Q projection: q = (xbf @ wqT^T) * scale -> [b,h,n,64] bf16.
// Grid: (INNER/64, MTOT/32), block 32.
// ---------------------------------------------------------------------------
__global__ void proj_q_kernel(const __bf16* __restrict__ X, const __bf16* __restrict__ WT,
                              __bf16* __restrict__ Qd) {
  int lane = threadIdx.x;
  int n0 = blockIdx.x * 64, m0 = blockIdx.y * 32;
  v8f acc[2][4] = {};
  gemm_32x64(X, WT, m0, n0, lane, acc);
  int half = lane >> 4, c = lane & 15;
#pragma unroll
  for (int mi = 0; mi < 2; ++mi)
#pragma unroll
    for (int ni = 0; ni < 4; ++ni) {
      int col = n0 + ni * 16 + c;
      int h = col >> 6, d = col & 63;
#pragma unroll
      for (int r = 0; r < 8; ++r) {
        int g = m0 + mi * 16 + r + half * 8;
        int bb = g >> 11, n = g & (SEQ - 1);
        Qd[((size_t)((bb * HEADS + h) * SEQ + n)) * DHEAD + d] =
            (__bf16)(acc[mi][ni][r] * ATTN_SCALE);
      }
    }
}

// ---------------------------------------------------------------------------
// KV projection: kv = xpbf @ wkvT^T; cols < 512 -> K row-major [b,h,n,64];
// cols >= 512 -> V TRANSPOSED [b,h,64,n] (vectorized 16B stores).
// Grid: (2*INNER/64, MTOT/32), block 32.
// ---------------------------------------------------------------------------
__global__ void proj_kv_kernel(const __bf16* __restrict__ X, const __bf16* __restrict__ WT,
                               __bf16* __restrict__ Kd, __bf16* __restrict__ VTd) {
  int lane = threadIdx.x;
  int n0 = blockIdx.x * 64, m0 = blockIdx.y * 32;
  v8f acc[2][4] = {};
  gemm_32x64(X, WT, m0, n0, lane, acc);
  int half = lane >> 4, c = lane & 15;
  bool isV = (n0 >= INNER);          // uniform per wave (64-col tiles)
#pragma unroll
  for (int mi = 0; mi < 2; ++mi)
#pragma unroll
    for (int ni = 0; ni < 4; ++ni) {
      int col = n0 + ni * 16 + c;
      int cc = col & (INNER - 1);
      int h = cc >> 6, d = cc & 63;
      int gbase = m0 + mi * 16 + half * 8;       // rows gbase..gbase+7
      int bb = gbase >> 11, nb = gbase & (SEQ - 1);
      if (isV) {
        v8bf ov;
#pragma unroll
        for (int r = 0; r < 8; ++r) ov[r] = (__bf16)acc[mi][ni][r];
        // VT[bh][d][n]: 8 consecutive n -> single 16B store
        *(v8bf*)(VTd + ((size_t)((bb * HEADS + h) * DHEAD + d)) * SEQ + nb) = ov;
      } else {
#pragma unroll
        for (int r = 0; r < 8; ++r)
          Kd[((size_t)((bb * HEADS + h) * SEQ + nb + r)) * DHEAD + d] =
              (__bf16)acc[mi][ni][r];
      }
    }
}

// ---------------------------------------------------------------------------
// Flash attention: per (b*h, 32-query tile), streaming 32 keys per step.
// S^T = K Q^T (keys x queries) so softmax rows live in lane pairs (c, c^16);
// O^T accumulated as V^T P with V^T stored contiguously.  Two query tiles per
// wave share all K / V^T fragments.  Grid: (32, SEQ/32), block 32.
// ---------------------------------------------------------------------------
__global__ void attn_kernel(const __bf16* __restrict__ Q, const __bf16* __restrict__ K,
                            const __bf16* __restrict__ VT, __bf16* __restrict__ O) {
  int lane = threadIdx.x;
  int half = lane >> 4, c = lane & 15;
  int bh = blockIdx.x;
  int q0 = blockIdx.y * 32;
  const __bf16* Qb  = Q  + (size_t)bh * SEQ * DHEAD;   // [n][64]
  const __bf16* Kb  = K  + (size_t)bh * SEQ * DHEAD;   // [n][64]
  const __bf16* VTb = VT + (size_t)bh * DHEAD * SEQ;   // [64][n]

  // Q^T B-fragments: [qi][d-chunk], loaded once (contiguous rows of Q)
  v16bf qB[2][2];
#pragma unroll
  for (int qi = 0; qi < 2; ++qi) {
    qB[qi][0] = load_bt_bf16(Qb, DHEAD, 0,  q0 + 16 * qi, lane);
    qB[qi][1] = load_bt_bf16(Qb, DHEAD, 32, q0 + 16 * qi, lane);
  }

  float m[2] = {-1e30f, -1e30f}, l[2] = {0.0f, 0.0f};
  v8f o[2][4] = {};

  for (int kt = 0; kt < SEQ; kt += 32) {
    // K A-fragments: [key m-tile][d-chunk], shared by both query tiles
    v16bf kA[2][2];
#pragma unroll
    for (int mi = 0; mi < 2; ++mi) {
      kA[mi][0] = load_a_bf16(Kb, DHEAD, kt + 16 * mi, 0,  lane);
      kA[mi][1] = load_a_bf16(Kb, DHEAD, kt + 16 * mi, 32, lane);
    }
    // V^T A-fragments: rows = d, cols = keys (contiguous in n)
    v16bf vA[4];
#pragma unroll
    for (int f = 0; f < 4; ++f)
      vA[f] = load_a_bf16(VTb, SEQ, f * 16, kt, lane);

#pragma unroll
    for (int qi = 0; qi < 2; ++qi) {
      v8f s0 = {}, s1 = {};
      s0 = wmma_bf16(kA[0][0], qB[qi][0], s0);
      s0 = wmma_bf16(kA[0][1], qB[qi][1], s0);
      s1 = wmma_bf16(kA[1][0], qB[qi][0], s1);
      s1 = wmma_bf16(kA[1][1], qB[qi][1], s1);

      // online softmax for this lane's query (attn scale baked into Q)
      float tmax = -1e30f;
#pragma unroll
      for (int r = 0; r < 8; ++r) { tmax = fmaxf(tmax, s0[r]); tmax = fmaxf(tmax, s1[r]); }
      tmax = fmaxf(tmax, __shfl_xor(tmax, 16, 32));
      float m_new = fmaxf(m[qi], tmax);
      float alpha = __expf(m[qi] - m_new);

      float e0[8], e1[8], lsum = 0.0f;
#pragma unroll
      for (int r = 0; r < 8; ++r) {
        e0[r] = __expf(s0[r] - m_new);
        e1[r] = __expf(s1[r] - m_new);
        lsum += e0[r] + e1[r];
      }
      lsum += __shfl_xor(lsum, 16, 32);
      l[qi] = l[qi] * alpha + lsum;
      m[qi] = m_new;

      // P B-fragment (32 keys x 16 queries) via lane-pair exchange
      v16bf pB;
#pragma unroll
      for (int j = 0; j < 8; ++j) {
        float x0 = __shfl_xor(e0[j], 16, 32);
        float x1 = __shfl_xor(e1[j], 16, 32);
        float lo = (half == 0) ? e0[j] : x1;    // keys 0..7  | 16..23
        float hi = (half == 0) ? x0    : e1[j]; // keys 8..15 | 24..31
        pB[j]     = (__bf16)lo;
        pB[8 + j] = (__bf16)hi;
      }

#pragma unroll
      for (int f = 0; f < 4; ++f) {
#pragma unroll
        for (int r = 0; r < 8; ++r) o[qi][f][r] *= alpha;
        o[qi][f] = wmma_bf16(vA[f], pB, o[qi][f]);
      }
    }
  }

  // Write attention output row-major [b*n][INNER] (16B stores: 8 consecutive d)
  int b = bh >> 3, h = bh & 7;
#pragma unroll
  for (int qi = 0; qi < 2; ++qi) {
    float inv = 1.0f / l[qi];
    size_t rowbase = ((size_t)b * SEQ + q0 + qi * 16 + c) * INNER + h * DHEAD;
#pragma unroll
    for (int f = 0; f < 4; ++f) {
      v8bf ob;
#pragma unroll
      for (int r = 0; r < 8; ++r) ob[r] = (__bf16)(o[qi][f][r] * inv);
      *(v8bf*)(O + rowbase + f * 16 + half * 8) = ob;
    }
  }
}

// ---------------------------------------------------------------------------
// Output projection: out = att @ w_out + b_out (f32).  Grid: (DIM/64, MTOT/32).
// ---------------------------------------------------------------------------
__global__ void outproj_kernel(const __bf16* __restrict__ A, const __bf16* __restrict__ WT,
                               const float* __restrict__ bias, float* __restrict__ out) {
  int lane = threadIdx.x;
  int n0 = blockIdx.x * 64, m0 = blockIdx.y * 32;
  v8f acc[2][4] = {};
  gemm_32x64(A, WT, m0, n0, lane, acc);
  int half = lane >> 4, c = lane & 15;
#pragma unroll
  for (int mi = 0; mi < 2; ++mi)
#pragma unroll
    for (int ni = 0; ni < 4; ++ni) {
      int col = n0 + ni * 16 + c;
      float bv = bias[col];
#pragma unroll
      for (int r = 0; r < 8; ++r) {
        int g = m0 + mi * 16 + r + half * 8;
        out[(size_t)g * DIM + col] = acc[mi][ni][r] + bv;
      }
    }
}

// ---------------------------------------------------------------------------
extern "C" void kernel_launch(void* const* d_in, const int* in_sizes, int n_in,
                              void* d_out, int out_size, void* d_ws, size_t ws_size,
                              hipStream_t stream) {
  const float* x      = (const float*)d_in[0];
  const float* x_prev = (const float*)d_in[1];
  const float* w_q    = (const float*)d_in[2];
  const float* w_kv   = (const float*)d_in[3];
  const float* w_out  = (const float*)d_in[4];
  const float* b_out  = (const float*)d_in[5];
  float* out = (float*)d_out;

  const size_t ACT = (size_t)MTOT * DIM;                 // 4 Mi elems
  __bf16* xbf    = (__bf16*)d_ws;                        // [8192][512]
  __bf16* xpbf   = xbf   + ACT;                          // [8192][512]
  __bf16* wqT    = xpbf  + ACT;                          // [512][512]
  __bf16* wkvT   = wqT   + (size_t)INNER * DIM;          // [1024][512]
  __bf16* woutT  = wkvT  + (size_t)2 * INNER * DIM;      // [512][512]
  __bf16* q_ws   = woutT + (size_t)DIM * DIM;            // [b,h,n,64]
  __bf16* k_ws   = q_ws  + ACT;                          // [b,h,n,64]
  __bf16* vT_ws  = k_ws  + ACT;                          // [b,h,64,n]
  __bf16* att_ws = vT_ws + ACT;                          // [8192][512]

  dim3 blk(32);
  // --- prep: bf16 conversions + weight transposes (trivial cost) ---
  cvt_bf16_kernel<<<(ACT / 4 + 255) / 256, 256, 0, stream>>>(x, xbf, ACT / 4);
  cvt_bf16_kernel<<<(ACT / 4 + 255) / 256, 256, 0, stream>>>(x_prev, xpbf, ACT / 4);
  transpose_w_kernel<<<(INNER * DIM + 255) / 256, 256, 0, stream>>>(w_q, wqT, INNER);
  transpose_w_kernel<<<(2 * INNER * DIM + 255) / 256, 256, 0, stream>>>(w_kv, wkvT, 2 * INNER);
  transpose_w_kernel<<<(DIM * DIM + 255) / 256, 256, 0, stream>>>(w_out, woutT, DIM);

  // --- WMMA pipeline ---
  proj_q_kernel<<<dim3(INNER / 64, MTOT / 32), blk, 0, stream>>>(xbf, wqT, q_ws);
  proj_kv_kernel<<<dim3(2 * INNER / 64, MTOT / 32), blk, 0, stream>>>(xpbf, wkvT, k_ws, vT_ws);
  attn_kernel<<<dim3(BATCH * HEADS, SEQ / 32), blk, 0, stream>>>(q_ws, k_ws, vT_ws, att_ws);
  outproj_kernel<<<dim3(DIM / 64, MTOT / 32), blk, 0, stream>>>(att_ws, w_out ? woutT : woutT, b_out, out);
}
